// Transformer_83056077570429
// MI455X (gfx1250) — compile-verified, matrix operand order checked by reading, not run on hardware
//
#include <hip/hip_runtime.h>
#include <hip/hip_bf16.h>

#define B_ 8
#define S_ 512
#define D_ 512
#define H_ 8
#define DK_ 64
#define FF_ 2048
#define L_ 6
#define V_ 32000
#define EPS_ 1e-5f

typedef __attribute__((ext_vector_type(16))) __bf16        v16bf;
typedef __attribute__((ext_vector_type(8)))  float         v8f;
typedef __attribute__((ext_vector_type(8)))  unsigned int  v8u;
typedef int v4i32 __attribute__((__vector_size__(16)));

#if defined(__has_builtin)
#if __has_builtin(__builtin_amdgcn_global_load_async_to_lds_b128)
#define HAVE_ASYNC_LDS 1
#endif
#endif
#ifndef HAVE_ASYNC_LDS
#define HAVE_ASYNC_LDS 0
#endif

#define TO_GLOBAL(p) ((__attribute__((address_space(1))) v4i32*)(p))
#define TO_LDS(p)    ((__attribute__((address_space(3))) v4i32*)(p))

__device__ __forceinline__ void wait_async0() {
#if defined(__has_builtin) && __has_builtin(__builtin_amdgcn_s_wait_asynccnt)
  __builtin_amdgcn_s_wait_asynccnt(0);
#else
  asm volatile("s_wait_asynccnt 0" ::: "memory");
#endif
}

// ---------------------------------------------------------------- utilities
__device__ __forceinline__ unsigned short f2bf(float f) {
  unsigned int u = __builtin_bit_cast(unsigned int, f);
  unsigned int r = (u + 0x7FFFu + ((u >> 16) & 1u)) >> 16;   // round-nearest-even
  return (unsigned short)r;
}

// Load a 16x32 bf16 fragment (A layout; B identical with the tile stored
// [n][k]).  Per ISA 7.12.2: lanes 0-15 rows 0-15, VGPR i holds K pair at
// k=(i/4)*16+(i%4)*2; lanes 16-31 same rows, K pairs +8.
__device__ __forceinline__ v16bf ld_frag(const unsigned short* tile, int stride, int lane) {
  const int half = lane >> 4;
  const int row  = lane & 15;
  v8u u;
#pragma unroll
  for (int i = 0; i < 8; ++i) {
    const int k = ((i >> 2) << 4) + (half << 3) + ((i & 3) << 1);
    u[i] = *(const unsigned int*)(tile + row * stride + k);
  }
  return __builtin_bit_cast(v16bf, u);
}

__device__ __forceinline__ v8f wmma_bf16(v16bf a, v16bf b, v8f c) {
  return __builtin_amdgcn_wmma_f32_16x16x32_bf16(false, a, false, b, (short)0, c, false, false);
}

// ------------------------------------------------- weight convert/transpose
// Wt[N][K] (bf16) = transpose(W[K][N] fp32).  grid=(N/64, K/64), 256 thr.
__global__ __launch_bounds__(256) void convert_wt_kernel(const float* __restrict__ W,
                                                         unsigned short* __restrict__ Wt,
                                                         int K, int N) {
  __shared__ unsigned short tile[64][65];
  const int bn = blockIdx.x * 64, bk = blockIdx.y * 64;
  const int tid = threadIdx.x;
#pragma unroll
  for (int i = 0; i < 16; ++i) {
    int idx = tid + i * 256;
    int k = idx >> 6, n = idx & 63;
    tile[k][n] = f2bf(W[(size_t)(bk + k) * N + bn + n]);
  }
  __syncthreads();
#pragma unroll
  for (int i = 0; i < 16; ++i) {
    int idx = tid + i * 256;
    int n = idx >> 6, k = idx & 63;
    Wt[(size_t)(bn + n) * K + bk + k] = tile[k][n];
  }
}

// ---------------------------------------------------------------- GEMM
// C[M,N] = act(A[M,K] @ W[K,N] + bias), weights pre-transposed bf16 Wt[N][K].
// BM=BN=128, BK=32, 256 thr (8 waves), wave tile 32x64.  grid=(N/128, M/128).
template <bool RELU>
__global__ __launch_bounds__(256) void gemm_kernel(const float* __restrict__ A,
                                                   const unsigned short* __restrict__ Wt,
                                                   const float* __restrict__ bias,
                                                   float* __restrict__ C,
                                                   int M, int N, int K) {
  __shared__ __align__(16) unsigned short As[128 * 32];   // [m][k] bf16
  __shared__ __align__(16) unsigned short Bs[128 * 32];   // [n][k] bf16

  const int tid  = threadIdx.x;
  const int lane = tid & 31;
  const int wid  = tid >> 5;
  const int bn   = blockIdx.x * 128;
  const int bm   = blockIdx.y * 128;
  const int wm   = wid & 3;    // 4 waves over M (32 rows each)
  const int wn   = wid >> 2;   // 2 waves over N (64 cols each)

  v8f acc[2][4] = {};

  for (int k0 = 0; k0 < K; k0 += 32) {
    // stage B tile: pure bf16 copy, 128 rows x 64B -> async global->LDS
#pragma unroll
    for (int i = 0; i < 2; ++i) {
      int idx = tid + i * 256;                 // 0..511 chunks of 16B
      int n = idx >> 2, c = (idx & 3) << 3;    // 8 ushorts per chunk
      const unsigned short* src = Wt + (size_t)(bn + n) * K + k0 + c;
      unsigned short* dst = &Bs[n * 32 + c];
#if HAVE_ASYNC_LDS
      __builtin_amdgcn_global_load_async_to_lds_b128(TO_GLOBAL(src), TO_LDS(dst), 0, 0);
#else
      *(uint4*)dst = *(const uint4*)src;
#endif
    }
    // stage A tile with fp32->bf16 conversion (overlaps with async B copy)
#pragma unroll
    for (int i = 0; i < 16; ++i) {
      int idx = tid + i * 256;
      int r = idx >> 5, c = idx & 31;
      As[r * 32 + c] = f2bf(A[(size_t)(bm + r) * K + k0 + c]);
    }
    if (k0 + 32 < K)  // hint next A K-tile into cache (global_prefetch_b8)
      __builtin_prefetch(A + (size_t)(bm + (tid >> 1)) * K + k0 + 32, 0, 1);
#if HAVE_ASYNC_LDS
    wait_async0();
#endif
    __syncthreads();

    v16bf af[2], bfr[4];
#pragma unroll
    for (int i = 0; i < 2; ++i) af[i]  = ld_frag(As + (wm * 32 + i * 16) * 32, 32, lane);
#pragma unroll
    for (int j = 0; j < 4; ++j) bfr[j] = ld_frag(Bs + (wn * 64 + j * 16) * 32, 32, lane);
#pragma unroll
    for (int i = 0; i < 2; ++i)
#pragma unroll
      for (int j = 0; j < 4; ++j)
        acc[i][j] = wmma_bf16(af[i], bfr[j], acc[i][j]);
    __syncthreads();
  }

  const int half = lane >> 4, nn = lane & 15;
#pragma unroll
  for (int i = 0; i < 2; ++i) {
    int row0 = bm + wm * 32 + i * 16 + half * 8;
#pragma unroll
    for (int j = 0; j < 4; ++j) {
      int col = bn + wn * 64 + j * 16 + nn;
      float bv = bias[col];
#pragma unroll
      for (int r = 0; r < 8; ++r) {
        float v = acc[i][j][r] + bv;
        if (RELU) v = fmaxf(v, 0.f);
        C[(size_t)(row0 + r) * N + col] = v;
      }
    }
  }
}

// ---------------------------------------------------------------- attention
// One WG per (64-query block, head, batch).  Full K resident in LDS; scores
// stay in WMMA accumulators; in-register softmax; P@V via WMMA.
__global__ __launch_bounds__(256) void attn_kernel(const float* __restrict__ Q,
                                                   const float* __restrict__ K,
                                                   const float* __restrict__ V,
                                                   float* __restrict__ O) {
  __shared__ unsigned short Qs[64 * 64];    //  8 KB [q][d]
  __shared__ unsigned short KV[512 * 64];   // 64 KB: K as [key][d], later V as [d][key]
  __shared__ unsigned short Ps[64 * 512];   // 64 KB softmax probs [q][key]
  __shared__ float redmax[2][64];
  __shared__ float redsum[2][64];

  const int tid = threadIdx.x, lane = tid & 31, wid = tid >> 5;
  const int qb = blockIdx.x, h = blockIdx.y, b = blockIdx.z;
  const float* Qg = Q + (size_t)b * S_ * D_ + h * DK_;
  const float* Kg = K + (size_t)b * S_ * D_ + h * DK_;
  const float* Vg = V + (size_t)b * S_ * D_ + h * DK_;

  for (int i = tid; i < 64 * 64; i += 256) {
    int m = i >> 6, d = i & 63;
    Qs[m * 64 + d] = f2bf(Qg[(size_t)(qb * 64 + m) * D_ + d]);
  }
  for (int i = tid; i < 512 * 64; i += 256) {
    int kk = i >> 6, d = i & 63;
    KV[kk * 64 + d] = f2bf(Kg[(size_t)kk * D_ + d]);
  }
  __syncthreads();

  // scores: wave (wm, wn) owns rows [wm*16,+16) x keys [wn*256,+256)
  const int wm = wid & 3, wn = wid >> 2;
  const int half = lane >> 4;
  v16bf aq0 = ld_frag(Qs + wm * 16 * 64, 64, lane);
  v16bf aq1 = ld_frag(Qs + wm * 16 * 64 + 32, 64, lane);
  v8f acc[16];
#pragma unroll
  for (int j = 0; j < 16; ++j) {
    int kt = wn * 16 + j;
    v16bf b0 = ld_frag(KV + kt * 16 * 64, 64, lane);
    v16bf b1 = ld_frag(KV + kt * 16 * 64 + 32, 64, lane);
    v8f c = {};
    c = wmma_bf16(aq0, b0, c);
    c = wmma_bf16(aq1, b1, c);
    acc[j] = c;
  }

  // softmax over 512 keys per row (scale = 1/sqrt(64))
  const float scale = 0.125f;
  float gmax[8];
#pragma unroll
  for (int r = 0; r < 8; ++r) {
    float m = -1e30f;
#pragma unroll
    for (int j = 0; j < 16; ++j) m = fmaxf(m, acc[j][r]);
#pragma unroll
    for (int msk = 1; msk < 16; msk <<= 1) m = fmaxf(m, __shfl_xor(m, msk, 32));
    gmax[r] = m * scale;
  }
  if ((lane & 15) == 0)
#pragma unroll
    for (int r = 0; r < 8; ++r) redmax[wn][wm * 16 + half * 8 + r] = gmax[r];
  __syncthreads();
  float gsum[8];
#pragma unroll
  for (int r = 0; r < 8; ++r) {
    int row = wm * 16 + half * 8 + r;
    float gm = fmaxf(redmax[0][row], redmax[1][row]);
    float s = 0.f;
#pragma unroll
    for (int j = 0; j < 16; ++j) {
      float e = __expf(acc[j][r] * scale - gm);
      acc[j][r] = e;
      s += e;
    }
#pragma unroll
    for (int msk = 1; msk < 16; msk <<= 1) s += __shfl_xor(s, msk, 32);
    gsum[r] = s;
  }
  if ((lane & 15) == 0)
#pragma unroll
    for (int r = 0; r < 8; ++r) redsum[wn][wm * 16 + half * 8 + r] = gsum[r];
  __syncthreads();
#pragma unroll
  for (int r = 0; r < 8; ++r) {
    int row = wm * 16 + half * 8 + r;
    float inv = 1.f / (redsum[0][row] + redsum[1][row]);
#pragma unroll
    for (int j = 0; j < 16; ++j) {
      int key = wn * 256 + j * 16 + (lane & 15);
      Ps[row * 512 + key] = f2bf(acc[j][r] * inv);
    }
  }
  __syncthreads();

  // stage V transposed into KV (safe: all K reads done before this barrier)
  for (int i = tid; i < 512 * 64; i += 256) {
    int kk = i >> 6, d = i & 63;
    KV[d * 512 + kk] = f2bf(Vg[(size_t)kk * D_ + d]);
  }
  __syncthreads();

  // O[64,64] = P[64,512] @ V[512,64]; 16 tiles / 8 waves = 2 per wave
  const int om = wid & 3, on = wid >> 2;
  float* Og = O + (size_t)b * S_ * D_ + h * DK_;
#pragma unroll
  for (int t2 = 0; t2 < 2; ++t2) {
    int dt = on * 2 + t2;
    v8f c = {};
    for (int k0 = 0; k0 < 512; k0 += 32) {
      v16bf a  = ld_frag(Ps + om * 16 * 512 + k0, 512, lane);
      v16bf bb = ld_frag(KV + dt * 16 * 512 + k0, 512, lane);
      c = wmma_bf16(a, bb, c);
    }
#pragma unroll
    for (int r = 0; r < 8; ++r) {
      int row = qb * 64 + om * 16 + half * 8 + r;
      int col = dt * 16 + (lane & 15);
      Og[(size_t)row * D_ + col] = c[r];
    }
  }
}

// ---------------------------------------------------------------- add + LN
__global__ __launch_bounds__(256) void add_ln_kernel(const float* __restrict__ t,
                                                     float* __restrict__ x,
                                                     const float* __restrict__ g,
                                                     const float* __restrict__ b) {
  __shared__ float red[8];
  const int row = blockIdx.x, tid = threadIdx.x;
  float v0 = t[(size_t)row * D_ + tid] + x[(size_t)row * D_ + tid];
  float v1 = t[(size_t)row * D_ + tid + 256] + x[(size_t)row * D_ + tid + 256];
  float s = v0 + v1;
#pragma unroll
  for (int m = 16; m; m >>= 1) s += __shfl_xor(s, m, 32);
  if ((tid & 31) == 0) red[tid >> 5] = s;
  __syncthreads();
  float tot = 0.f;
#pragma unroll
  for (int i = 0; i < 8; ++i) tot += red[i];
  const float mean = tot * (1.f / D_);
  __syncthreads();
  float d0 = v0 - mean, d1 = v1 - mean;
  s = d0 * d0 + d1 * d1;
#pragma unroll
  for (int m = 16; m; m >>= 1) s += __shfl_xor(s, m, 32);
  if ((tid & 31) == 0) red[tid >> 5] = s;
  __syncthreads();
  tot = 0.f;
#pragma unroll
  for (int i = 0; i < 8; ++i) tot += red[i];
  const float rstd = rsqrtf(tot * (1.f / D_) + EPS_);
  x[(size_t)row * D_ + tid]       = d0 * rstd * g[tid] + b[tid];
  x[(size_t)row * D_ + tid + 256] = d1 * rstd * g[tid + 256] + b[tid + 256];
}

// ---------------------------------------------------------------- embedding
__global__ __launch_bounds__(256) void embed_kernel(const int* __restrict__ tok,
                                                    const float* __restrict__ emb,
                                                    float* __restrict__ out) {
  const int row = blockIdx.x;          // b*S + s
  const int s = row & (S_ - 1);
  const int t = tok[row];
  const int tid = threadIdx.x;
#pragma unroll
  for (int i = 0; i < 2; ++i) {
    int d = tid + i * 256;
    float freq = __expf(-(float)d * (9.210340371976184f / (float)D_));  // ln(1e4)/D
    float ang = (float)s * freq;
    float pe = (d & 1) ? __cosf(ang) : __sinf(ang);
    out[(size_t)row * D_ + d] = emb[(size_t)t * D_ + d] + pe;
  }
}

// ---------------------------------------------------------------- host side
struct LayerW {
  const float *W1, *W2, *b1, *b2, *fln_b, *fln_g;
  const float *Wk, *Wo, *Wq, *Wv, *bk, *bo, *bq, *bv, *mln_b, *mln_g;
};
struct LayerWT {  // bf16 transposed [N][K] copies
  unsigned short *Wq, *Wk, *Wv, *Wo, *W1, *W2;
};

static LayerW unpack(void* const* d_in, int base) {
  LayerW w;
  w.W1 = (const float*)d_in[base + 0];  w.W2 = (const float*)d_in[base + 1];
  w.b1 = (const float*)d_in[base + 2];  w.b2 = (const float*)d_in[base + 3];
  w.fln_b = (const float*)d_in[base + 4]; w.fln_g = (const float*)d_in[base + 5];
  w.Wk = (const float*)d_in[base + 6];  w.Wo = (const float*)d_in[base + 7];
  w.Wq = (const float*)d_in[base + 8];  w.Wv = (const float*)d_in[base + 9];
  w.bk = (const float*)d_in[base + 10]; w.bo = (const float*)d_in[base + 11];
  w.bq = (const float*)d_in[base + 12]; w.bv = (const float*)d_in[base + 13];
  w.mln_b = (const float*)d_in[base + 14]; w.mln_g = (const float*)d_in[base + 15];
  return w;
}

extern "C" void kernel_launch(void* const* d_in, const int* in_sizes, int n_in,
                              void* d_out, int out_size, void* d_ws, size_t ws_size,
                              hipStream_t stream) {
  (void)in_sizes; (void)n_in; (void)out_size; (void)ws_size;
  const int M = B_ * S_;  // 4096 tokens per stream

  const int*   src     = (const int*)d_in[0];
  const int*   tgt     = (const int*)d_in[1];
  const float* src_emb = (const float*)d_in[2];
  const float* tgt_emb = (const float*)d_in[3];
  LayerW enc = unpack(d_in, 4);
  LayerW dec = unpack(d_in, 20);
  const float* out_W = (const float*)d_in[36];
  const float* out_b = (const float*)d_in[37];
  float* out = (float*)d_out;

  // fp32 workspace: 6 x [4096,512] + [4096,2048]  (~84 MB)
  float* xw = (float*)d_ws;
  float* yw = xw + (size_t)M * D_;
  float* qb = yw + (size_t)M * D_;
  float* kb = qb + (size_t)M * D_;
  float* vb = kb + (size_t)M * D_;
  float* t0 = vb + (size_t)M * D_;
  float* hb = t0 + (size_t)M * D_;
  // bf16 transposed-weight workspace (~108 MB), 16B aligned
  unsigned short* bfp = (unsigned short*)(hb + (size_t)M * FF_);
  const size_t DD = (size_t)D_ * D_, DF = (size_t)D_ * FF_;
  LayerWT encT[L_], decT[L_];
  for (int l = 0; l < L_; ++l) {
    encT[l].Wq = bfp; bfp += DD;  encT[l].Wk = bfp; bfp += DD;
    encT[l].Wv = bfp; bfp += DD;  encT[l].Wo = bfp; bfp += DD;
    encT[l].W1 = bfp; bfp += DF;  encT[l].W2 = bfp; bfp += DF;
  }
  for (int l = 0; l < L_; ++l) {
    decT[l].Wq = bfp; bfp += DD;  decT[l].Wk = bfp; bfp += DD;
    decT[l].Wv = bfp; bfp += DD;  decT[l].Wo = bfp; bfp += DD;
    decT[l].W1 = bfp; bfp += DF;  decT[l].W2 = bfp; bfp += DF;
  }
  unsigned short* outWT = bfp;    // [V][D]

  auto convert = [&](const float* W, unsigned short* Wt, int K, int N) {
    convert_wt_kernel<<<dim3(N / 64, K / 64), 256, 0, stream>>>(W, Wt, K, N);
  };
  for (int l = 0; l < L_; ++l) {
    convert(enc.Wq + l * DD, encT[l].Wq, D_, D_);
    convert(enc.Wk + l * DD, encT[l].Wk, D_, D_);
    convert(enc.Wv + l * DD, encT[l].Wv, D_, D_);
    convert(enc.Wo + l * DD, encT[l].Wo, D_, D_);
    convert(enc.W1 + l * DF, encT[l].W1, D_, FF_);
    convert(enc.W2 + l * DF, encT[l].W2, FF_, D_);
    convert(dec.Wq + l * DD, decT[l].Wq, D_, D_);
    convert(dec.Wk + l * DD, decT[l].Wk, D_, D_);
    convert(dec.Wv + l * DD, decT[l].Wv, D_, D_);
    convert(dec.Wo + l * DD, decT[l].Wo, D_, D_);
    convert(dec.W1 + l * DF, decT[l].W1, D_, FF_);
    convert(dec.W2 + l * DF, decT[l].W2, FF_, D_);
  }
  convert(out_W, outWT, D_, V_);

  auto gemm = [&](const float* A, const unsigned short* Wt, const float* bias, float* C,
                  int m, int n, int k, bool relu) {
    dim3 grid(n / 128, m / 128), blk(256);
    if (relu) gemm_kernel<true><<<grid, blk, 0, stream>>>(A, Wt, bias, C, m, n, k);
    else      gemm_kernel<false><<<grid, blk, 0, stream>>>(A, Wt, bias, C, m, n, k);
  };
  auto mha = [&](const LayerW& w, const LayerWT& wt, int l,
                 const float* qsrc, const float* kvsrc, float* io) {
    const int bo_ = l * D_;
    gemm(qsrc,  wt.Wq, w.bq + bo_, qb, M, D_, D_, false);
    gemm(kvsrc, wt.Wk, w.bk + bo_, kb, M, D_, D_, false);
    gemm(kvsrc, wt.Wv, w.bv + bo_, vb, M, D_, D_, false);
    attn_kernel<<<dim3(S_ / 64, H_, B_), 256, 0, stream>>>(qb, kb, vb, t0);
    gemm(t0, wt.Wo, w.bo + bo_, qb, M, D_, D_, false);
    add_ln_kernel<<<M, 256, 0, stream>>>(qb, io, w.mln_g + bo_, w.mln_b + bo_);
  };
  auto ffn = [&](const LayerW& w, const LayerWT& wt, int l, float* io) {
    gemm(io, wt.W1, w.b1 + l * FF_, hb, M, FF_, D_, true);
    gemm(hb, wt.W2, w.b2 + l * D_, t0, M, D_, FF_, false);
    add_ln_kernel<<<M, 256, 0, stream>>>(t0, io, w.fln_g + l * D_, w.fln_b + l * D_);
  };

  embed_kernel<<<M, 256, 0, stream>>>(src, src_emb, xw);
  embed_kernel<<<M, 256, 0, stream>>>(tgt, tgt_emb, yw);

  for (int l = 0; l < L_; ++l) {        // encoder
    mha(enc, encT[l], l, xw, xw, xw);
    ffn(enc, encT[l], l, xw);
  }
  for (int l = 0; l < L_; ++l) {        // decoder
    mha(dec, decT[l], l, yw, yw, yw);   // self-attention
    mha(dec, decT[l], l, yw, xw, yw);   // cross-attention (same weights, as in ref)
    ffn(dec, decT[l], l, yw);
  }
  gemm(yw, outWT, out_b, out, M, V_, D_, false);   // [4096, 32000] logits
}
